// PointTransformerLayer_3049426780496
// MI455X (gfx1250) — matrix operand order repaired and checked by reading, not compiled, not run on hardware
//
#include <hip/hip_runtime.h>
#include <hip/hip_bf16.h>

// ---- problem constants (match reference) ----
constexpr int B_  = 2;
constexpr int N_  = 20000;
constexpr int C_  = 64;
constexpr int K_  = 16;
constexpr int CS_ = 16;
constexpr float EPS_ = 1e-5f;
constexpr int NT_ = N_ / 16;           // 1250 point-tiles

typedef __attribute__((ext_vector_type(16))) __bf16 v16bf;
typedef __attribute__((ext_vector_type(8)))  float  v8f;

// =====================================================================
// Kernel 1: the three 1x1-conv projections as bf16 WMMA GEMMs.
//   xq = wq*q + bq ; xk = wk*k + bk ; xv = wv*q + bv   (all [B,C,N])
// One wave computes a 16(outC) x 16(points) tile, K=64 -> 2 WMMAs.
// =====================================================================
__global__ void __launch_bounds__(256)
pt_proj_kernel(const float* __restrict__ q, const float* __restrict__ k,
               const float* __restrict__ wq, const float* __restrict__ bq,
               const float* __restrict__ wk, const float* __restrict__ bk,
               const float* __restrict__ wv, const float* __restrict__ bv,
               float* __restrict__ xq, float* __restrict__ xk, float* __restrict__ xv)
{
    const int wave = (blockIdx.x * blockDim.x + threadIdx.x) >> 5;
    const int lid  = threadIdx.x & 31;
    const int wavesPerProj = B_ * 4 * NT_;           // 4 outC tiles of 16
    const int proj = wave / wavesPerProj;
    if (proj >= 3) return;
    int r = wave % wavesPerProj;
    const int b  = r / (4 * NT_);  r %= 4 * NT_;
    const int ct = r / NT_;                          // out-channel tile
    const int nt = r % NT_;                          // point tile

    const float *W, *bias, *X;
    float *O;
    if (proj == 0)      { W = wq; bias = bq; X = q; O = xq; }
    else if (proj == 1) { W = wk; bias = bk; X = k; O = xk; }
    else                { W = wv; bias = bv; X = q; O = xv; }

    const int l = lid & 15;          // A: row M ; B: column ; D: column
    const int h = lid >> 4;          // half-wave selector
    const int n0 = nt * 16;

    v8f acc = {};
    #pragma unroll
    for (int cc = 0; cc < 2; ++cc) {                 // K = cc*32 .. cc*32+31
        // A-matrix: 16x32 bf16, lane(l,h): elems 0..7 -> K=cc*32+h*8+j,
        //                                 elems 8..15 -> K=cc*32+16+h*8+j
        v16bf a;
        const float* wrow = W + (ct * 16 + l) * C_ + cc * 32;
        #pragma unroll
        for (int j = 0; j < 8; ++j) {
            a[j]     = (__bf16)wrow[h * 8 + j];
            a[8 + j] = (__bf16)wrow[16 + h * 8 + j];
        }
        // B-matrix: 32x16 bf16, lane(l,h): elems 0..7 -> K=cc*32+h*8+j, elems 8..15 -> K=cc*32+16+h*8+(j-8), col=l
        v16bf bm;
        const float* xcol = X + (size_t)b * C_ * N_ + n0 + l;
        #pragma unroll
        for (int j = 0; j < 16; ++j)
            bm[j] = (__bf16)xcol[(size_t)(cc * 32 + (j < 8 ? h * 8 + j : 16 + h * 8 + (j - 8))) * N_];

        acc = __builtin_amdgcn_wmma_f32_16x16x32_bf16(
                  false, a, false, bm, (short)0, acc, false, false);
    }
    // D layout: VGPR rr, lanes 0-15: M=rr, lanes 16-31: M=8+rr ; col = l
    #pragma unroll
    for (int rr = 0; rr < 8; ++rr) {
        const int row = ct * 16 + rr + h * 8;
        O[((size_t)b * C_ + row) * N_ + n0 + l] = acc[rr] + bias[row];
    }
}

// =====================================================================
// Kernel 2: fused attention. One wave per point (b,n).
//   - positional MLP inner vector per neighbor (lanes 0-15) -> LDS
//   - 64x16 "w" feature matrix built in B-matrix layout, 2 WMMAs (w1)
//   - BN2+ReLU, in-register repack, K-padded WMMA (w2)
//   - softmax over K via 16-lane shfl_xor butterflies
//   - grouped weighted sum over (x_v + p_r)
// All LDS is wave-private (in-order per wave) -> only wave_barrier fences.
// =====================================================================
__global__ void __launch_bounds__(256)
pt_attn_kernel(const float* __restrict__ coord, const int* __restrict__ nbr,
               const float* __restrict__ xq, const float* __restrict__ xkp,
               const float* __restrict__ xvp,
               const float* __restrict__ p1_w, const float* __restrict__ p1_b,
               const float* __restrict__ pg, const float* __restrict__ pb,
               const float* __restrict__ pm, const float* __restrict__ pv,
               const float* __restrict__ p2_w, const float* __restrict__ p2_b,
               const float* __restrict__ g1, const float* __restrict__ b1,
               const float* __restrict__ m1, const float* __restrict__ v1,
               const float* __restrict__ w1_w, const float* __restrict__ w1_b,
               const float* __restrict__ g2, const float* __restrict__ b2,
               const float* __restrict__ m2, const float* __restrict__ v2,
               const float* __restrict__ w2_w, const float* __restrict__ w2_b,
               float* __restrict__ out)
{
    __shared__ int   idx_s[8][K_];        // neighbor indices
    __shared__ float t_s[8][K_ * 3];      // positional-MLP hidden (3 per nbr)
    __shared__ float xq_s[8][C_];         // query features of this point
    __shared__ float p_s[8][CS_ * K_];    // softmaxed attention probs [g][k]

    const int wvid = threadIdx.x >> 5;
    const int lid  = threadIdx.x & 31;
    const int pid  = blockIdx.x * (blockDim.x >> 5) + wvid;
    if (pid >= B_ * N_) return;           // wave-uniform (B_*N_ % 8 == 0 anyway)
    const int b = pid / N_;
    const int n = pid % N_;
    const int kk = lid & 15;              // neighbor owned in WMMA phase
    const int h  = lid >> 4;              // channel-half selector

    // ---- phase 0: stage x_q[:,n] and per-neighbor positional hidden ----
    xq_s[wvid][lid]      = xq[((size_t)b * C_ + lid) * N_ + n];
    xq_s[wvid][lid + 32] = xq[((size_t)b * C_ + lid + 32) * N_ + n];

    if (lid < 16) {
        const int id = nbr[((size_t)b * N_ + n) * K_ + lid];
        idx_s[wvid][lid] = id;
        const float c0 = coord[((size_t)b * N_ + id) * 3 + 0];
        const float c1 = coord[((size_t)b * N_ + id) * 3 + 1];
        const float c2 = coord[((size_t)b * N_ + id) * 3 + 2];
        #pragma unroll
        for (int i = 0; i < 3; ++i) {
            float y = p1_w[i * 3 + 0] * c0 + p1_w[i * 3 + 1] * c1
                    + p1_w[i * 3 + 2] * c2 + p1_b[i];
            y = (y - pm[i]) * (pg[i] * rsqrtf(pv[i] + EPS_)) + pb[i];
            t_s[wvid][lid * 3 + i] = fmaxf(y, 0.0f);
        }
    }
    __builtin_amdgcn_wave_barrier();

    // ---- phase 1: build w = relu(bn1(x_k - x_q + p_r)) in B-matrix layout ----
    const int   id = idx_s[wvid][kk];
    const float t0 = t_s[wvid][kk * 3 + 0];
    const float t1 = t_s[wvid][kk * 3 + 1];
    const float t2 = t_s[wvid][kk * 3 + 2];

    v16bf bmat0, bmat1;
    #pragma unroll
    for (int cc = 0; cc < 2; ++cc) {
        #pragma unroll
        for (int j = 0; j < 16; ++j) {
            const int c = cc * 32 + (j < 8 ? h * 8 + j : 16 + h * 8 + (j - 8));
            const float xkv = xkp[((size_t)b * C_ + c) * N_ + id];
            const float pr  = p2_w[c * 3 + 0] * t0 + p2_w[c * 3 + 1] * t1
                            + p2_w[c * 3 + 2] * t2 + p2_b[c];
            float wf = xkv - xq_s[wvid][c] + pr;
            wf = (wf - m1[c]) * (g1[c] * rsqrtf(v1[c] + EPS_)) + b1[c];
            wf = fmaxf(wf, 0.0f);
            if (cc == 0) bmat0[j] = (__bf16)wf; else bmat1[j] = (__bf16)wf;
        }
    }
    v8f acc = {};
    #pragma unroll
    for (int cc = 0; cc < 2; ++cc) {
        v16bf amat;
        const float* wrow = w1_w + (lid & 15) * C_ + cc * 32;
        #pragma unroll
        for (int j = 0; j < 8; ++j) {
            amat[j]     = (__bf16)wrow[h * 8 + j];
            amat[8 + j] = (__bf16)wrow[16 + h * 8 + j];
        }
        acc = __builtin_amdgcn_wmma_f32_16x16x32_bf16(
                  false, amat, false, (cc == 0 ? bmat0 : bmat1),
                  (short)0, acc, false, false);
    }

    // ---- phase 2: + w1_b, BN2, ReLU (D layout: row g = r + h*8) ----
    float u[8];
    #pragma unroll
    for (int r = 0; r < 8; ++r) {
        const int g = r + h * 8;
        float x = acc[r] + w1_b[g];
        x = (x - m2[g]) * (g2[g] * rsqrtf(v2[g] + EPS_)) + b2[g];
        u[r] = fmaxf(x, 0.0f);
    }

    // ---- phase 3: w2 as K-padded (16->32) WMMA; D regs copied to B directly ----
    v16bf a2, b2v;
    #pragma unroll
    for (int j = 0; j < 8; ++j) {
        const float other = __shfl(u[j], kk + 16, 32);   // rows g=8..15 of col kk
        b2v[j]     = (__bf16)u[j];
        b2v[8 + j] = (__bf16)0.0f;
        a2[j]      = (__bf16)w2_w[(lid & 15) * CS_ + h * 8 + j];
        a2[8 + j]  = (__bf16)0.0f;                       // K padding
    }
    v8f zc = {};
    v8f acc2 = __builtin_amdgcn_wmma_f32_16x16x32_bf16(
                   false, a2, false, b2v, (short)0, zc, false, false);

    // ---- phase 4: + w2_b, softmax over K (16 lanes of same half) ----
    #pragma unroll
    for (int r = 0; r < 8; ++r) {
        const int g = r + h * 8;
        float x = acc2[r] + w2_b[g];
        float mx = x;
        #pragma unroll
        for (int msk = 1; msk < 16; msk <<= 1)
            mx = fmaxf(mx, __shfl_xor(mx, msk, 32));
        const float e = __expf(x - mx);
        float s = e;
        #pragma unroll
        for (int msk = 1; msk < 16; msk <<= 1)
            s += __shfl_xor(s, msk, 32);
        p_s[wvid][g * 16 + kk] = e / s;
    }
    __builtin_amdgcn_wave_barrier();

    // ---- phase 5: grouped weighted sum: out[c] = sum_k (x_v+p_r)[c,k]*p[c&15,k]
    const int c0v = lid, c1v = lid + 32;     // c0&15 == c1&15 == lid&15
    float a0 = 0.0f, a1 = 0.0f;
    #pragma unroll
    for (int k2 = 0; k2 < 16; ++k2) {
        const int   id2 = idx_s[wvid][k2];
        const float s0  = t_s[wvid][k2 * 3 + 0];
        const float s1  = t_s[wvid][k2 * 3 + 1];
        const float s2  = t_s[wvid][k2 * 3 + 2];
        const float pw  = p_s[wvid][(lid & 15) * 16 + k2];
        const float pr0 = p2_w[c0v * 3 + 0] * s0 + p2_w[c0v * 3 + 1] * s1
                        + p2_w[c0v * 3 + 2] * s2 + p2_b[c0v];
        const float pr1 = p2_w[c1v * 3 + 0] * s0 + p2_w[c1v * 3 + 1] * s1
                        + p2_w[c1v * 3 + 2] * s2 + p2_b[c1v];
        a0 += (xvp[((size_t)b * C_ + c0v) * N_ + id2] + pr0) * pw;
        a1 += (xvp[((size_t)b * C_ + c1v) * N_ + id2] + pr1) * pw;
    }
    out[((size_t)b * C_ + c0v) * N_ + n] = a0;
    out[((size_t)b * C_ + c1v) * N_ + n] = a1;
}

extern "C" void kernel_launch(void* const* d_in, const int* in_sizes, int n_in,
                              void* d_out, int out_size, void* d_ws, size_t ws_size,
                              hipStream_t stream) {
    const float* coord = (const float*)d_in[0];
    const float* q     = (const float*)d_in[1];
    const float* k     = (const float*)d_in[2];
    const int*   nbr   = (const int*)  d_in[3];
    const float* wq    = (const float*)d_in[4];
    const float* bq    = (const float*)d_in[5];
    const float* wk    = (const float*)d_in[6];
    const float* bk    = (const float*)d_in[7];
    const float* wv    = (const float*)d_in[8];
    const float* bv    = (const float*)d_in[9];
    const float* p1_w  = (const float*)d_in[10];
    const float* p1_b  = (const float*)d_in[11];
    const float* pbn_g = (const float*)d_in[12];
    const float* pbn_b = (const float*)d_in[13];
    const float* pbn_m = (const float*)d_in[14];
    const float* pbn_v = (const float*)d_in[15];
    const float* p2_w  = (const float*)d_in[16];
    const float* p2_b  = (const float*)d_in[17];
    const float* g1    = (const float*)d_in[18];
    const float* b1    = (const float*)d_in[19];
    const float* m1    = (const float*)d_in[20];
    const float* v1    = (const float*)d_in[21];
    const float* w1_w  = (const float*)d_in[22];
    const float* w1_b  = (const float*)d_in[23];
    const float* g2    = (const float*)d_in[24];
    const float* b2    = (const float*)d_in[25];
    const float* m2    = (const float*)d_in[26];
    const float* v2    = (const float*)d_in[27];
    const float* w2_w  = (const float*)d_in[28];
    const float* w2_b  = (const float*)d_in[29];
    float* out = (float*)d_out;

    const size_t plane = (size_t)B_ * C_ * N_;       // 2.56M floats
    float* xq = (float*)d_ws;
    float* xk = xq + plane;
    float* xv = xk + plane;

    // Kernel 1: 3 projections * B * 4 outC-tiles * 1250 point-tiles = 30000 waves
    {
        const int waves = 3 * B_ * 4 * NT_;
        dim3 grid(waves / 8), block(256);
        pt_proj_kernel<<<grid, block, 0, stream>>>(q, k, wq, bq, wk, bk, wv, bv,
                                                   xq, xk, xv);
    }
    // Kernel 2: one wave per point, B*N = 40000 waves
    {
        const int waves = B_ * N_;
        dim3 grid(waves / 8), block(256);
        pt_attn_kernel<<<grid, block, 0, stream>>>(coord, nbr, xq, xk, xv,
                                                   p1_w, p1_b, pbn_g, pbn_b, pbn_m, pbn_v,
                                                   p2_w, p2_b,
                                                   g1, b1, m1, v1, w1_w, w1_b,
                                                   g2, b2, m2, v2, w2_w, w2_b,
                                                   out);
    }
}
